// GKA_14087492730930
// MI455X (gfx1250) — matrix operand order repaired and checked by reading, not compiled, not run on hardware
//
#include <hip/hip_runtime.h>
#include <hip/hip_bf16.h>
#include <math.h>

typedef _Float16 h16;
typedef __attribute__((ext_vector_type(16))) _Float16 v16h;
typedef __attribute__((ext_vector_type(8)))  _Float16 v8h;
typedef __attribute__((ext_vector_type(8)))  float    v8f;

#define NB   8
#define NC   1024
#define NT   1024
#define DX   3
#define DY   32
#define HD   128

// workspace layout (bytes, all 16B aligned)
#define OFF_W0T  0          // [128][32]  f16, k<3 real (K padded to 32)
#define OFF_W1T  8192       // [128][128] f16, n-major (K contiguous)
#define OFF_W2T  40960      // [128][128] f16
#define OFF_W3T  73728      // [16][128]  f16, n<3 real (N padded to 16)
#define OFF_SIG  77824      // [8192][4]  f32

static __device__ __forceinline__ v8f wmma_f16(v16h a, v16h b, v8f c) {
  // D = A(16x32 f16) * B(32x16 f16) + C(16x16 f32)
  return __builtin_amdgcn_wmma_f32_16x16x32_f16(false, a, false, b, (short)0, c,
                                                false, false);
}
static __device__ __forceinline__ v16h cat8(v8h a, v8h b) {
  v16h r;
#pragma unroll
  for (int i = 0; i < 8; ++i) { r[i] = a[i]; r[i + 8] = b[i]; }
  return r;
}
static __device__ __forceinline__ v16h zero16h() {
  v16h r;
#pragma unroll
  for (int i = 0; i < 16; ++i) r[i] = (h16)0.f;
  return r;
}
static __device__ __forceinline__ v8f zero8f() {
  v8f r;
#pragma unroll
  for (int i = 0; i < 8; ++i) r[i] = 0.f;
  return r;
}
// wave-private LDS store->load ordering (DS ops are in-order per wave)
static __device__ __forceinline__ void lds_fence() {
  asm volatile("s_wait_dscnt 0" ::: "memory");
}

// ---------------------------------------------------------------------------
// Kernel 1: convert weights f32 -> f16, transposed (n-major, K contiguous)
// ---------------------------------------------------------------------------
__global__ void GKA_prep(const float* __restrict__ W0, const float* __restrict__ W1,
                         const float* __restrict__ W2, const float* __restrict__ W3,
                         h16* __restrict__ W0T, h16* __restrict__ W1T,
                         h16* __restrict__ W2T, h16* __restrict__ W3T) {
  int i = blockIdx.x * 256 + threadIdx.x;
  if (i < 128 * 32) {
    int n = i >> 5, k = i & 31;
    W0T[n * 32 + k] = (k < DX) ? (h16)W0[k * HD + n] : (h16)0.f;
  } else if (i < 128 * 32 + 128 * 128) {
    int j = i - 128 * 32, n = j >> 7, k = j & 127;
    W1T[n * HD + k] = (h16)W1[k * HD + n];
  } else if (i < 128 * 32 + 2 * 128 * 128) {
    int j = i - (128 * 32 + 128 * 128), n = j >> 7, k = j & 127;
    W2T[n * HD + k] = (h16)W2[k * HD + n];
  } else if (i < 128 * 32 + 2 * 128 * 128 + 16 * 128) {
    int j = i - (128 * 32 + 2 * 128 * 128), n = j >> 7, k = j & 127;
    W3T[n * HD + k] = (n < DX) ? (h16)W3[k * DX + n] : (h16)0.f;
  }
}

// ---------------------------------------------------------------------------
// Kernel 2: sig = exp(MLP(xt)) for all 8192 target points.
// 64 blocks x 256 threads (8 waves); each wave owns 16 rows (wave-private
// LDS slice -> no block barriers needed, only per-wave DS fences).
// ---------------------------------------------------------------------------
__global__ void __launch_bounds__(256) GKA_mlp(
    const float* __restrict__ xt,
    const h16* __restrict__ W0T, const h16* __restrict__ W1T,
    const h16* __restrict__ W2T, const h16* __restrict__ W3T,
    const float* __restrict__ b0, const float* __restrict__ b1,
    const float* __restrict__ b2, const float* __restrict__ b3,
    float* __restrict__ sig) {
  __shared__ __attribute__((aligned(16))) h16 X[128][136];  // activations (f16)

  const int tid = threadIdx.x;
  const int w = tid >> 5, lane = tid & 31, lo = lane & 15, hi = lane >> 4;
  const int R0 = blockIdx.x * 128;  // global row base (row = b*NT + t)
  const int rw = w * 16;            // wave's local row base

  // ---- layer 0: (16x3 padded to 16x32) x (32x128), one WMMA per tile ----
  v16h a0 = zero16h();
  if (hi == 0) {
    const float* xr = xt + (size_t)(R0 + rw + lo) * DX;
    a0[0] = (h16)xr[0]; a0[1] = (h16)xr[1]; a0[2] = (h16)xr[2];
  }
  v8f acc[8];
#pragma unroll
  for (int nt = 0; nt < 8; ++nt) {
    float bv = b0[nt * 16 + lo];
    v8f c;
#pragma unroll
    for (int r = 0; r < 8; ++r) c[r] = bv;
    const h16* bp = W0T + (nt * 16 + lo) * 32 + hi * 16;
    c = wmma_f16(a0, cat8(*(const v8h*)bp, *(const v8h*)(bp + 8)), c);
#pragma unroll
    for (int r = 0; r < 8; ++r) c[r] = fmaxf(c[r], 0.f);
    acc[nt] = c;
  }
#pragma unroll
  for (int nt = 0; nt < 8; ++nt)
#pragma unroll
    for (int r = 0; r < 8; ++r)
      X[rw + r + 8 * hi][nt * 16 + lo] = (h16)acc[nt][r];
  lds_fence();

  // ---- layers 1,2: (16x128) x (128x128), 4 WMMA per tile ----
#pragma unroll 1
  for (int layer = 0; layer < 2; ++layer) {
    const h16* Wt = layer ? W2T : W1T;
    const float* bb = layer ? b2 : b1;
    v16h a[4];
#pragma unroll
    for (int q = 0; q < 4; ++q) {
      const h16* ap = &X[rw + lo][q * 32 + hi * 16];
      a[q] = cat8(*(const v8h*)ap, *(const v8h*)(ap + 8));
    }
#pragma unroll
    for (int nt = 0; nt < 8; ++nt) {
      float bv = bb[nt * 16 + lo];
      v8f c;
#pragma unroll
      for (int r = 0; r < 8; ++r) c[r] = bv;
#pragma unroll
      for (int q = 0; q < 4; ++q) {
        const h16* bp = Wt + (nt * 16 + lo) * HD + q * 32 + hi * 16;
        c = wmma_f16(a[q], cat8(*(const v8h*)bp, *(const v8h*)(bp + 8)), c);
      }
#pragma unroll
      for (int r = 0; r < 8; ++r) c[r] = fmaxf(c[r], 0.f);
      acc[nt] = c;
    }
#pragma unroll
    for (int nt = 0; nt < 8; ++nt)
#pragma unroll
      for (int r = 0; r < 8; ++r)
        X[rw + r + 8 * hi][nt * 16 + lo] = (h16)acc[nt][r];
    lds_fence();
  }

  // ---- layer 3: (16x128) x (128x3 padded to 128x16), sig = exp(.) ----
  {
    v16h a[4];
#pragma unroll
    for (int q = 0; q < 4; ++q) {
      const h16* ap = &X[rw + lo][q * 32 + hi * 16];
      a[q] = cat8(*(const v8h*)ap, *(const v8h*)(ap + 8));
    }
    float bv = (lo < DX) ? b3[lo] : 0.f;
    v8f c;
#pragma unroll
    for (int r = 0; r < 8; ++r) c[r] = bv;
#pragma unroll
    for (int q = 0; q < 4; ++q) {
      const h16* bp = W3T + lo * HD + q * 32 + hi * 16;
      c = wmma_f16(a[q], cat8(*(const v8h*)bp, *(const v8h*)(bp + 8)), c);
    }
    if (lo < DX) {
#pragma unroll
      for (int r = 0; r < 8; ++r) {
        int row = R0 + rw + r + 8 * hi;
        sig[(size_t)row * 4 + lo] = __expf(c[r]);
      }
    }
  }
}

// ---------------------------------------------------------------------------
// Kernel 3: fused P-GEMM + online softmax + S*yc (flash-attention style).
// Grid (8 target-tiles, 8 batches) x 256 threads; wave w owns 16 targets.
// P = f(targets) . g(contexts), rank-7 features padded to K=32.
// ---------------------------------------------------------------------------
__global__ void __launch_bounds__(256) GKA_attn(
    const float* __restrict__ xc, const float* __restrict__ yc,
    const float* __restrict__ xt, const float* __restrict__ sig,
    float* __restrict__ out) {
  __shared__ __attribute__((aligned(16))) h16 fT[128][8];   // target features
  __shared__ __attribute__((aligned(16))) h16 gT[32][8];    // context features
  __shared__ __attribute__((aligned(16))) h16 ycT[32][40];  // ycT[dy][c] chunk
  __shared__ __attribute__((aligned(16))) h16 Sld[8][16][40];  // per-wave S

  const int tid = threadIdx.x;
  const int w = tid >> 5, lane = tid & 31, lo = lane & 15, hi = lane >> 4;
  const int b = blockIdx.y;
  const int T0 = blockIdx.x * 128;

  // target features f = [2*sig*xt, -sig, -sum(sig*xt^2), 0]
  if (tid < 128) {
    size_t base = (size_t)b * NT + (T0 + tid);
    float s0 = sig[base * 4 + 0], s1 = sig[base * 4 + 1], s2 = sig[base * 4 + 2];
    const float* xr = xt + base * DX;
    float x0 = xr[0], x1 = xr[1], x2 = xr[2];
    fT[tid][0] = (h16)(2.f * s0 * x0);
    fT[tid][1] = (h16)(2.f * s1 * x1);
    fT[tid][2] = (h16)(2.f * s2 * x2);
    fT[tid][3] = (h16)(-s0);
    fT[tid][4] = (h16)(-s1);
    fT[tid][5] = (h16)(-s2);
    fT[tid][6] = (h16)(-(s0 * x0 * x0 + s1 * x1 * x1 + s2 * x2 * x2));
    fT[tid][7] = (h16)0.f;
  }
  __syncthreads();

  v16h aP = zero16h();  // A fragment: 16 targets x K=32 (K<8 real)
  if (hi == 0) {
    v8h f8 = *(const v8h*)&fT[w * 16 + lo][0];
#pragma unroll
    for (int i = 0; i < 8; ++i) aP[i] = f8[i];
  }

  v8f acc0 = zero8f(), acc1 = zero8f();
  float mrow[8], lrow[8];
#pragma unroll
  for (int r = 0; r < 8; ++r) { mrow[r] = -3.0e38f; lrow[r] = 0.f; }

#pragma unroll 1
  for (int c0 = 0; c0 < NC; c0 += 32) {
    __syncthreads();  // protect gT/ycT reuse across iterations
    if (tid < 32) {   // context features g = [x, x^2, 1, 0]
      const float* xr = xc + ((size_t)b * NC + c0 + tid) * DX;
      float x0 = xr[0], x1 = xr[1], x2 = xr[2];
      gT[tid][0] = (h16)x0;        gT[tid][1] = (h16)x1;
      gT[tid][2] = (h16)x2;        gT[tid][3] = (h16)(x0 * x0);
      gT[tid][4] = (h16)(x1 * x1); gT[tid][5] = (h16)(x2 * x2);
      gT[tid][6] = (h16)1.f;       gT[tid][7] = (h16)0.f;
    }
#pragma unroll
    for (int k = 0; k < 4; ++k) {  // yc chunk, transposed to dy-major f16
      int e = tid * 4 + k, ci = e >> 5, dy = e & 31;
      ycT[dy][ci] = (h16)yc[((size_t)b * NC + c0 + ci) * DY + dy];
    }
    if (c0 + 32 < NC)  // global_prefetch_b8 of next yc chunk
      __builtin_prefetch(yc + ((size_t)b * NC + c0 + 32) * DY, 0, 0);
    __syncthreads();

    // P tile: 16 targets x 32 contexts via two feature WMMAs
    v8f P[2];
#pragma unroll
    for (int nt = 0; nt < 2; ++nt) {
      v16h bg = zero16h();
      if (hi == 0) {
        v8h g8 = *(const v8h*)&gT[nt * 16 + lo][0];
#pragma unroll
        for (int i = 0; i < 8; ++i) bg[i] = g8[i];
      }
      P[nt] = wmma_f16(aP, bg, zero8f());
    }

    // online softmax over the context (N) direction, stats in f32
    v8f e0, e1;
#pragma unroll
    for (int r = 0; r < 8; ++r) {
      float mc = fmaxf(P[0][r], P[1][r]);
      mc = fmaxf(mc, __shfl_xor(mc, 1));
      mc = fmaxf(mc, __shfl_xor(mc, 2));
      mc = fmaxf(mc, __shfl_xor(mc, 4));
      mc = fmaxf(mc, __shfl_xor(mc, 8));
      float mn = fmaxf(mrow[r], mc);
      float sc = __expf(mrow[r] - mn);
      lrow[r] *= sc; acc0[r] *= sc; acc1[r] *= sc;
      float ea = __expf(P[0][r] - mn);
      float eb = __expf(P[1][r] - mn);
      e0[r] = ea; e1[r] = eb;
      float ls = ea + eb;
      ls += __shfl_xor(ls, 1); ls += __shfl_xor(ls, 2);
      ls += __shfl_xor(ls, 4); ls += __shfl_xor(ls, 8);
      lrow[r] += ls;
      mrow[r] = mn;
    }

    // re-layout S (D fragment -> A fragment) through wave-private LDS
#pragma unroll
    for (int r = 0; r < 8; ++r) {
      Sld[w][r + 8 * hi][lo]      = (h16)e0[r];
      Sld[w][r + 8 * hi][lo + 16] = (h16)e1[r];
    }
    lds_fence();
    const h16* sp = &Sld[w][lo][0];
    v16h aS = cat8(*(const v8h*)(sp + 16 * hi), *(const v8h*)(sp + 16 * hi + 8));

    // acc += S (16x32) * yc (32x32), two WMMAs
    {
      const h16* yp0 = &ycT[lo][hi * 16];
      acc0 = wmma_f16(aS, cat8(*(const v8h*)yp0, *(const v8h*)(yp0 + 8)), acc0);
      const h16* yp1 = &ycT[16 + lo][hi * 16];
      acc1 = wmma_f16(aS, cat8(*(const v8h*)yp1, *(const v8h*)(yp1 + 8)), acc1);
    }
  }

  // finalize: out[b,t,dy] = acc / l
#pragma unroll
  for (int r = 0; r < 8; ++r) {
    float inv = 1.f / lrow[r];
    int t = T0 + w * 16 + r + 8 * hi;
    float* op = out + ((size_t)b * NT + t) * DY;
    op[lo]      = acc0[r] * inv;
    op[lo + 16] = acc1[r] * inv;
  }
}

// ---------------------------------------------------------------------------
extern "C" void kernel_launch(void* const* d_in, const int* in_sizes, int n_in,
                              void* d_out, int out_size, void* d_ws, size_t ws_size,
                              hipStream_t stream) {
  const float* xc = (const float*)d_in[0];
  const float* yc = (const float*)d_in[1];
  const float* xt = (const float*)d_in[2];
  const float* W0 = (const float*)d_in[3];
  const float* b0 = (const float*)d_in[4];
  const float* W1 = (const float*)d_in[5];
  const float* b1 = (const float*)d_in[6];
  const float* W2 = (const float*)d_in[7];
  const float* b2 = (const float*)d_in[8];
  const float* W3 = (const float*)d_in[9];
  const float* b3 = (const float*)d_in[10];
  float* out = (float*)d_out;

  char* ws = (char*)d_ws;
  h16* W0T = (h16*)(ws + OFF_W0T);
  h16* W1T = (h16*)(ws + OFF_W1T);
  h16* W2T = (h16*)(ws + OFF_W2T);
  h16* W3T = (h16*)(ws + OFF_W3T);
  float* sig = (float*)(ws + OFF_SIG);

  GKA_prep<<<152, 256, 0, stream>>>(W0, W1, W2, W3, W0T, W1T, W2T, W3T);
  GKA_mlp<<<64, 256, 0, stream>>>(xt, W0T, W1T, W2T, W3T, b0, b1, b2, b3, sig);
  GKA_attn<<<dim3(8, 8), 256, 0, stream>>>(xc, yc, xt, sig, out);
  (void)in_sizes; (void)n_in; (void)out_size; (void)ws_size;
}
